// NonLocalBlock_82944408420677
// MI455X (gfx1250) — matrix operand order repaired and checked by reading, not compiled, not run on hardware
//
#include <hip/hip_runtime.h>

#define DEVI __device__ __forceinline__

typedef __bf16 bf16;
typedef __attribute__((ext_vector_type(16))) __bf16 v16bf;
typedef __attribute__((ext_vector_type(8)))  float  v8f;

static constexpr int Bn = 8, C = 256, IC = 128, N = 4096;

// ---- helpers -------------------------------------------------------------

DEVI bf16 tobf(float f) {
  union { float f; unsigned u; } v; v.f = f;
  unsigned r = (v.u + 0x7FFFu + ((v.u >> 16) & 1u)) >> 16;   // RNE f32->bf16
  union { unsigned short s; bf16 b; } o; o.s = (unsigned short)r;
  return o.b;
}

// A-fragment (16x32 bf16), row-major source, leading dim ld (elements).
// lanes 0-15: row M, K 0..15; lanes 16-31: K 16..31.
DEVI v16bf load_afrag(const bf16* base, int ld, int row0, int k0, int lane) {
  int r    = row0 + (lane & 15);
  int koff = (lane >> 4) * 16;
  const bf16* p = base + (size_t)r * ld + k0 + koff;
  v16bf a;
  ((float4*)&a)[0] = *(const float4*)(p);
  ((float4*)&a)[1] = *(const float4*)(p + 8);
  return a;
}

// B-fragment (32x16 bf16) from a [col][k] tile (K contiguous per column).
// lanes 0-15: K=0..15, lanes 16-31: K=16..31, column = lane&15.
DEVI v16bf load_bfrag(const bf16* base, int ld, int col0, int k0, int lane) {
  int c  = col0 + (lane & 15);
  int kb = k0 + (lane >> 4) * 16;
  const bf16* p = base + (size_t)c * ld + kb;
  v16bf b;
  ((float4*)&b)[0] = *(const float4*)(p);
  ((float4*)&b)[1] = *(const float4*)(p + 8);
  return b;
}

DEVI v8f wmma_bf16(v16bf a, v16bf b, v8f c) {
  return __builtin_amdgcn_wmma_f32_16x16x32_bf16(false, a, false, b, (short)0, c,
                                                 false, false);
}

// ---- kernel 0: weights f32 -> bf16 (vectorized, 4 elems/thread) ----------

__global__ void k0_cvt_weights(const float* __restrict__ wphi,
                               const float* __restrict__ wtheta,
                               const float* __restrict__ wg,
                               const float* __restrict__ wmask,
                               bf16* __restrict__ ophi, bf16* __restrict__ otheta,
                               bf16* __restrict__ og,   bf16* __restrict__ omask) {
  const int SZ4 = IC * C / 4;                // 8192 float4 chunks per array
  int i = blockIdx.x * blockDim.x + threadIdx.x;
  int which = i / SZ4, r = (i % SZ4) * 4;
  const float* src = which == 0 ? wphi : which == 1 ? wtheta
                   : which == 2 ? wg : wmask;
  bf16* dst = which == 0 ? ophi : which == 1 ? otheta
            : which == 2 ? og : omask;
  float4 d = *(const float4*)(src + r);
  __align__(8) bf16 tmp[4] = {tobf(d.x), tobf(d.y), tobf(d.z), tobf(d.w)};
  *(float2*)(dst + r) = *(const float2*)tmp;
}

// ---- kernel 1: projection GEMM  W[ic,C] * x[C,N] -------------------------
// transpose_out==0: dst = [ic][N] (phi); else dst = [N][ic] (theta, g)

__global__ void __launch_bounds__(256)
k1_proj(const float* __restrict__ x, const bf16* __restrict__ wbf,
        bf16* __restrict__ dst, int transpose_out) {
  __shared__ __align__(16) bf16 lds_x[64 * 40];   // [j][k], ld=40
  const int b  = blockIdx.y;
  const int j0 = blockIdx.x * 64;
  const int t = threadIdx.x, lane = t & 31, w = t >> 5;

  v8f acc[4];
  const v8f vz = {0.f,0.f,0.f,0.f,0.f,0.f,0.f,0.f};
  for (int js = 0; js < 4; ++js) acc[js] = vz;

  for (int kt = 0; kt < C / 32; ++kt) {
    const int k0 = kt * 32;
    __syncthreads();
    // 32k x 64j tile: float4 along j, transpose into [j][k]
    for (int c = t; c < 512; c += 256) {
      int kk = c >> 4, j4 = (c & 15) * 4;
      float4 d = *(const float4*)(x + ((size_t)(b * C + k0 + kk)) * N + j0 + j4);
      lds_x[(j4 + 0) * 40 + kk] = tobf(d.x);
      lds_x[(j4 + 1) * 40 + kk] = tobf(d.y);
      lds_x[(j4 + 2) * 40 + kk] = tobf(d.z);
      lds_x[(j4 + 3) * 40 + kk] = tobf(d.w);
    }
    __syncthreads();
    v16bf afr = load_afrag(wbf, C, 16 * w, k0, lane);
    for (int js = 0; js < 4; ++js) {
      v16bf bfr = load_bfrag(lds_x, 40, 16 * js, 0, lane);
      acc[js] = wmma_bf16(afr, bfr, acc[js]);
    }
  }

  const int mr0 = (lane >> 4) * 8;
  if (!transpose_out) {
    for (int js = 0; js < 4; ++js) {
      int col = j0 + 16 * js + (lane & 15);
      for (int r = 0; r < 8; ++r)
        dst[((size_t)b * IC + 16 * w + mr0 + r) * N + col] = tobf(acc[js][r]);
    }
  } else {
    for (int js = 0; js < 4; ++js) {
      int col = j0 + 16 * js + (lane & 15);
      __align__(16) bf16 tmp[8];
      for (int r = 0; r < 8; ++r) tmp[r] = tobf(acc[js][r]);
      *(float4*)(&dst[((size_t)b * N + col) * IC + 16 * w + mr0]) =
          *(const float4*)tmp;
    }
  }
}

// ---- kernel 2: column softmax stats (softmax over axis=1 == rows i) ------

__global__ void __launch_bounds__(256)
k2_stats(const bf16* __restrict__ theta, const bf16* __restrict__ phi,
         float* __restrict__ colmax, float* __restrict__ colsum) {
  __shared__ __align__(16) bf16 lds_phi[128 * 136];   // [j][ic], ld=136
  const int b  = blockIdx.y;
  const int j0 = blockIdx.x * 128;
  const int t = threadIdx.x, lane = t & 31, w = t >> 5;
  const bf16* phib   = phi   + (size_t)b * IC * N;
  const bf16* thetab = theta + (size_t)b * N * IC;

  // stage phi panel once: b128 along j, scatter-transpose into [j][ic]
  for (int c = t; c < 2048; c += 256) {
    int ic = c >> 4, j8 = (c & 15) * 8;
    float4 d = *(const float4*)(phib + (size_t)ic * N + j0 + j8);
    const bf16* e = (const bf16*)&d;
    for (int r = 0; r < 8; ++r) lds_phi[(j8 + r) * 136 + ic] = e[r];
  }
  __syncthreads();

  // hoist the 4 invariant B-fragments for this wave's 16 columns
  const int jw = 16 * w;
  v16bf bfr[4];
  for (int ks = 0; ks < 4; ++ks)
    bfr[ks] = load_bfrag(lds_phi, 136, jw, 32 * ks, lane);

  float m_run = -3.0e38f, l_run = 0.f;
  for (int it = 0; it < N / 16; ++it) {
    const int i0 = it * 16;
    v8f acc = {0.f,0.f,0.f,0.f,0.f,0.f,0.f,0.f};
    for (int ks = 0; ks < 4; ++ks) {
      v16bf a = load_afrag(thetab, IC, i0, 32 * ks, lane);
      acc = wmma_bf16(a, bfr[ks], acc);
    }
    float tmax = acc[0];
    for (int r = 1; r < 8; ++r) tmax = fmaxf(tmax, acc[r]);
    tmax = fmaxf(tmax, __shfl_xor(tmax, 16, 32));   // merge rows 0-7 / 8-15
    float m_new = fmaxf(m_run, tmax);
    float scale = __expf(m_run - m_new);
    float ps = 0.f;
    for (int r = 0; r < 8; ++r) ps += __expf(acc[r] - m_new);
    ps += __shfl_xor(ps, 16, 32);
    l_run = l_run * scale + ps;
    m_run = m_new;
  }
  if (lane < 16) {
    colmax[(size_t)b * N + j0 + jw + lane] = m_run;
    colsum[(size_t)b * N + j0 + jw + lane] = l_run;
  }
}

// ---- kernel 3: recompute scores -> p = exp(s-cm)/cs -> y = p*g -----------
// Block = 32 i rows x 64 m tile. All 8 waves produce one 16x16 score subtile
// (isub = w&1, msub = w>>1), then all 8 waves run p*g (wave w owns o-block w).

__global__ void __launch_bounds__(256)
k3_attn_y(const bf16* __restrict__ theta, const bf16* __restrict__ phi,
          const bf16* __restrict__ g, const float* __restrict__ colmax,
          const float* __restrict__ colsum, bf16* __restrict__ y) {
  __shared__ __align__(16) bf16 lds_phi[64 * 136];  // [m][ic], ld=136
  __shared__ __align__(16) bf16 lds_g[128 * 72];    // [o][m],  ld=72
  __shared__ __align__(16) bf16 lds_p[32 * 72];     // [i][m],  ld=72
  const int b  = blockIdx.y;
  const int i0 = blockIdx.x * 32;
  const int t = threadIdx.x, lane = t & 31, w = t >> 5;
  const bf16* phib   = phi   + (size_t)b * IC * N;
  const bf16* thetab = theta + (size_t)b * N * IC;
  const bf16* gb     = g     + (size_t)b * N * IC;
  const int isub = w & 1, msub = w >> 1;

  v16bf ath[4];
  for (int ks = 0; ks < 4; ++ks)
    ath[ks] = load_afrag(thetab, IC, i0 + 16 * isub, 32 * ks, lane);

  const v8f vz = {0.f,0.f,0.f,0.f,0.f,0.f,0.f,0.f};
  v8f yacc[2] = {vz, vz};

  for (int mt = 0; mt < N / 64; ++mt) {
    const int m0 = mt * 64;
    // phi [ic][m] -> [m][ic]: b128 along m (8 elems), scatter-transpose
    for (int c = t; c < 1024; c += 256) {
      int ic = c >> 3, m8 = (c & 7) * 8;
      float4 d = *(const float4*)(phib + (size_t)ic * N + m0 + m8);
      const bf16* e = (const bf16*)&d;
      for (int r = 0; r < 8; ++r) lds_phi[(m8 + r) * 136 + ic] = e[r];
    }
    // g [m][o] -> [o][m]: b128 along o (8 elems), scatter-transpose
    for (int c = t; c < 1024; c += 256) {
      int mm = c >> 4, o8 = (c & 15) * 8;
      float4 d = *(const float4*)(gb + (size_t)(m0 + mm) * IC + o8);
      const bf16* e = (const bf16*)&d;
      for (int r = 0; r < 8; ++r) lds_g[(o8 + r) * 72 + mm] = e[r];
    }
    if (mt + 1 < N / 64)                    // warm L2 for the next phi tile
      __builtin_prefetch(phib + (size_t)(t & 127) * N + m0 + 64, 0, 1);
    __syncthreads();

    // scores: each wave one 16x16 subtile, K=128
    {
      const int mcol = m0 + 16 * msub + (lane & 15);
      float cm  = colmax[(size_t)b * N + mcol];
      float rcs = 1.0f / colsum[(size_t)b * N + mcol];
      v8f s = vz;
      for (int ks = 0; ks < 4; ++ks) {
        v16bf bb = load_bfrag(lds_phi, 136, 16 * msub, 32 * ks, lane);
        s = wmma_bf16(ath[ks], bb, s);
      }
      const int mr0 = (lane >> 4) * 8;
      for (int r = 0; r < 8; ++r) {
        float p = __expf(s[r] - cm) * rcs;
        lds_p[(16 * isub + mr0 + r) * 72 + 16 * msub + (lane & 15)] = tobf(p);
      }
    }
    __syncthreads();

    // y += p * g : wave w owns o columns [16w,16w+16), both i subtiles, K=64
    for (int is2 = 0; is2 < 2; ++is2)
      for (int k2 = 0; k2 < 2; ++k2) {
        v16bf ap = load_afrag(lds_p, 72, 16 * is2, 32 * k2, lane);
        v16bf bb = load_bfrag(lds_g, 72, 16 * w, 32 * k2, lane);
        yacc[is2] = wmma_bf16(ap, bb, yacc[is2]);
      }
    __syncthreads();
  }

  const int mr0 = (lane >> 4) * 8;
  for (int is2 = 0; is2 < 2; ++is2) {
    int ocol = 16 * w + (lane & 15);
    for (int r = 0; r < 8; ++r) {
      int row = i0 + 16 * is2 + mr0 + r;
      y[((size_t)b * N + row) * IC + ocol] = tobf(yacc[is2][r]);
    }
  }
}

// ---- kernel 4: out = w_mask[C,IC] * y^T + x3 -----------------------------

__global__ void __launch_bounds__(256)
k4_mask_out(const bf16* __restrict__ y, const bf16* __restrict__ wmask,
            const float* __restrict__ x3, float* __restrict__ out) {
  __shared__ __align__(16) bf16 lds_y[64 * 136];   // [j][o], ld=136
  const int b  = blockIdx.y;
  const int j0 = blockIdx.x * 64;
  const int t = threadIdx.x, lane = t & 31, w = t >> 5;
  const bf16* yb = y + (size_t)b * N * IC;

  // direct copy, b128 global -> b128 LDS
  for (int c = t; c < 1024; c += 256) {
    int jj = c >> 4, o8 = (c & 15) * 8;
    *(float4*)&lds_y[jj * 136 + o8] =
        *(const float4*)(yb + (size_t)(j0 + jj) * IC + o8);
  }
  __syncthreads();

  for (int csub = 0; csub < 2; ++csub) {
    const int crow0 = 32 * w + 16 * csub;
    v16bf afr[4];
    for (int ks = 0; ks < 4; ++ks)
      afr[ks] = load_afrag(wmask, IC, crow0, 32 * ks, lane);
    for (int js = 0; js < 4; ++js) {
      v8f acc = {0.f,0.f,0.f,0.f,0.f,0.f,0.f,0.f};
      for (int ks = 0; ks < 4; ++ks) {
        v16bf bb = load_bfrag(lds_y, 136, 16 * js, 32 * ks, lane);
        acc = wmma_bf16(afr[ks], bb, acc);
      }
      const int col = j0 + 16 * js + (lane & 15);
      const int mr0 = (lane >> 4) * 8;
      for (int r = 0; r < 8; ++r) {
        size_t o = ((size_t)b * C + crow0 + mr0 + r) * N + col;
        out[o] = acc[r] + x3[o];
      }
    }
  }
}

// ---- host launch ---------------------------------------------------------

extern "C" void kernel_launch(void* const* d_in, const int* in_sizes, int n_in,
                              void* d_out, int out_size, void* d_ws, size_t ws_size,
                              hipStream_t stream) {
  (void)in_sizes; (void)n_in; (void)out_size; (void)ws_size;
  const float* x1     = (const float*)d_in[0];
  const float* x2     = (const float*)d_in[1];
  const float* x3     = (const float*)d_in[2];
  const float* wphi   = (const float*)d_in[3];
  const float* wtheta = (const float*)d_in[4];
  const float* wg     = (const float*)d_in[5];
  const float* wmask  = (const float*)d_in[6];
  float* out = (float*)d_out;

  char* ws = (char*)d_ws;
  size_t off = 0;
  auto alloc = [&](size_t bytes) -> char* {
    char* p = ws + off;
    off += (bytes + 255) & ~(size_t)255;
    return p;
  };
  bf16* wphi_b   = (bf16*)alloc((size_t)IC * C * 2);
  bf16* wtheta_b = (bf16*)alloc((size_t)IC * C * 2);
  bf16* wg_b     = (bf16*)alloc((size_t)IC * C * 2);
  bf16* wmask_b  = (bf16*)alloc((size_t)C * IC * 2);
  bf16* phi_b    = (bf16*)alloc((size_t)Bn * IC * N * 2);   // [b][ic][n]
  bf16* theta_b  = (bf16*)alloc((size_t)Bn * N * IC * 2);   // [b][n][ic]
  bf16* g_b      = (bf16*)alloc((size_t)Bn * N * IC * 2);   // [b][n][ic]
  bf16* y_b      = (bf16*)alloc((size_t)Bn * N * IC * 2);   // [b][n][ic]
  float* colmax  = (float*)alloc((size_t)Bn * N * 4);
  float* colsum  = (float*)alloc((size_t)Bn * N * 4);

  k0_cvt_weights<<<dim3(4 * IC * C / 4 / 256), 256, 0, stream>>>(
      wphi, wtheta, wg, wmask, wphi_b, wtheta_b, wg_b, wmask_b);

  k1_proj<<<dim3(N / 64, Bn), 256, 0, stream>>>(x1, wphi_b,   phi_b,   0);
  k1_proj<<<dim3(N / 64, Bn), 256, 0, stream>>>(x2, wtheta_b, theta_b, 1);
  k1_proj<<<dim3(N / 64, Bn), 256, 0, stream>>>(x3, wg_b,     g_b,     1);

  k2_stats<<<dim3(N / 128, Bn), 256, 0, stream>>>(theta_b, phi_b, colmax, colsum);

  k3_attn_y<<<dim3(N / 32, Bn), 256, 0, stream>>>(theta_b, phi_b, g_b,
                                                  colmax, colsum, y_b);

  k4_mask_out<<<dim3(N / 64, Bn), 256, 0, stream>>>(y_b, wmask_b, x3, out);
}
